// LNSeqTagModel_2757369004625
// MI455X (gfx1250) — matrix operand order repaired and checked by reading, not compiled, not run on hardware
//
#include <hip/hip_runtime.h>
#include <hip/hip_bf16.h>

typedef __bf16 bf16;
typedef __attribute__((ext_vector_type(16))) __bf16 v16bf;
typedef __attribute__((ext_vector_type(8)))  __bf16 v8bf;
typedef __attribute__((ext_vector_type(8)))  float  v8f;

#define B_  64
#define T_  256
#define I_  512
#define H_  512
#define G4  2048
#define O_  512
#define BH  (B_*H_)
#define LDSW_ELEM 520   // padded Whh row stride in bf16 elems (1040B -> 4-bank shift/row)

__device__ __forceinline__ float sigm(float x){ return 1.f/(1.f + __expf(-x)); }
__device__ __forceinline__ float tanh_f(float x){ return 2.f/(1.f + __expf(-2.f*x)) - 1.f; }

// A-matrix 16x32 bf16 fragment: lane l holds row l%16;
// elems 0..15 = K[kb+half .. +15], half = (l>=16)?16:0
__device__ __forceinline__ v16bf load_a(const bf16* rowp, int kb, int lane){
  int half = (lane & 16) ? 16 : 0;
  v8bf lo = *(const v8bf*)(rowp + kb + half);
  v8bf hi = *(const v8bf*)(rowp + kb + half + 8);
  v16bf a;
#pragma unroll
  for (int i = 0; i < 8; ++i){ a[i] = lo[i]; a[i+8] = hi[i]; }
  return a;
}

// B-matrix 32x16 bf16 fragment from a [N,K] row-major slab (global, 32B-aligned rows)
__device__ __forceinline__ v16bf load_b(const bf16* rowp, int kb, int lane){
  int kk = kb + ((lane & 16) ? 16 : 0);
  return *(const v16bf*)(rowp + kk);
}

// Same fragment but from a 16B-aligned (padded) LDS row: two b128 loads.
__device__ __forceinline__ v16bf load_b_lds(const bf16* rowp, int kb, int lane){
  int kk = kb + ((lane & 16) ? 16 : 0);
  v8bf lo = *(const v8bf*)(rowp + kk);
  v8bf hi = *(const v8bf*)(rowp + kk + 8);
  v16bf b;
#pragma unroll
  for (int i = 0; i < 8; ++i){ b[i] = lo[i]; b[i+8] = hi[i]; }
  return b;
}

__device__ __forceinline__ v8f wmma_bf16(v16bf a, v16bf b, v8f c){
  return __builtin_amdgcn_wmma_f32_16x16x32_bf16(false, a, false, b, (short)0, c, false, false);
}

// ---------------- prep kernels ----------------
__global__ __launch_bounds__(256) void k_cvt(const float* __restrict__ s, bf16* __restrict__ d, int n){
  int i = blockIdx.x * 256 + threadIdx.x;
  if (i < n) d[i] = (bf16)s[i];
}

__global__ __launch_bounds__(256) void k_h0(const float* __restrict__ h0f, const float* __restrict__ h0b,
                                            bf16* __restrict__ hsf, bf16* __restrict__ hsb){
  int i = blockIdx.x * 256 + threadIdx.x;   // 0..BH-1, slot 0 of hs
  int j = i & (H_ - 1);
  hsf[i] = (bf16)h0f[j];
  hsb[i] = (bf16)h0b[j];
}

__global__ void k_bar(unsigned* bar){ if (threadIdx.x < 8) bar[threadIdx.x] = 0u; }

// ---------------- phase 1: xg = X @ Wih^T + bih + bhh  (per direction) ----------------
// grid = 256 t-slots * 16 n-blocks; block tile 64(batch) x 128(N); 8 waves x 4 tiles
__global__ __launch_bounds__(256) void k_xg(const bf16* __restrict__ Xbf, const bf16* __restrict__ Wih,
                                            const float* __restrict__ bih, const float* __restrict__ bhh,
                                            float* __restrict__ xg, int reverse){
  int tslot = blockIdx.x >> 4;
  int n0 = (blockIdx.x & 15) * 128;
  int tx = reverse ? (T_ - 1 - tslot) : tslot;
  int w = threadIdx.x >> 5, lane = threadIdx.x & 31;
  int mt  = (w & 3) * 16;
  int ntb = (w >> 2) * 4;
  int col = lane & 15;

  const bf16* arow = Xbf + ((size_t)(mt + col) * T_ + tx) * I_;  // X stays [B,T,I]
  const bf16* brow[4];
#pragma unroll
  for (int i = 0; i < 4; ++i)
    brow[i] = Wih + (size_t)(n0 + (ntb + i) * 16 + col) * I_;

  v8f acc[4] = {};
#pragma unroll 4
  for (int kb = 0; kb < I_; kb += 32){
    v16bf a = load_a(arow, kb, lane);
#pragma unroll
    for (int i = 0; i < 4; ++i){
      v16bf b = load_b(brow[i], kb, lane);
      acc[i] = wmma_bf16(a, b, acc[i]);
    }
  }
  int rb = mt + ((lane & 16) ? 8 : 0);
#pragma unroll
  for (int i = 0; i < 4; ++i){
    int n = n0 + (ntb + i) * 16 + col;
    float bias = bih[n] + bhh[n];
#pragma unroll
    for (int r = 0; r < 8; ++r){
      int bm = rb + r;
      xg[((size_t)tslot * B_ + bm) * G4 + n] = acc[i][r] + bias;
    }
  }
}

// ---------------- grid-wide barrier (device-scope atomics) ----------------
__device__ __forceinline__ void gsync(unsigned* cnt, unsigned* gen, unsigned nb){
  __threadfence();
  __syncthreads();
  if (threadIdx.x == 0){
    unsigned g = __atomic_load_n(gen, __ATOMIC_RELAXED);
    unsigned a = __atomic_fetch_add(cnt, 1u, __ATOMIC_ACQ_REL) + 1;
    if (a == nb){
      __atomic_store_n(cnt, 0u, __ATOMIC_RELAXED);
      __atomic_fetch_add(gen, 1u, __ATOMIC_RELEASE);
    } else {
      while (__atomic_load_n(gen, __ATOMIC_ACQUIRE) == g)
        __builtin_amdgcn_s_sleep(1);
    }
  }
  __syncthreads();
  __threadfence();
}

// ---------------- phase 2: persistent bidirectional scan ----------------
// 32 blocks: blocks 0..15 = forward (hidden slice j0=32*blk..+32), 16..31 = backward.
// Whh slice (128 x 512 bf16) is staged ONCE into LDS via async global->LDS copies
// (gfx1250 ASYNCcnt path), then every timestep's recurrent GEMM reads B from LDS.
__global__ __launch_bounds__(256) void k_scan(const bf16* __restrict__ Whh_f, const bf16* __restrict__ Whh_b,
                                              const float* __restrict__ xg_f, const float* __restrict__ xg_b,
                                              const float* __restrict__ c0_f, const float* __restrict__ c0_b,
                                              bf16* hs_f, bf16* hs_b, unsigned* bar){
  __shared__ bf16  lds_w[128 * LDSW_ELEM];   // ~130KB: block's Whh slice, padded rows
  __shared__ float lds_g[B_][132];
  __shared__ float lds_c[B_][32];
  int dir = blockIdx.x >> 4;
  int j0  = (blockIdx.x & 15) * 32;
  const bf16*  Whh = dir ? Whh_b : Whh_f;
  const float* xg  = dir ? xg_b  : xg_f;
  const float* c0  = dir ? c0_b  : c0_f;
  bf16*        hs  = dir ? hs_b  : hs_f;

  int w = threadIdx.x >> 5, lane = threadIdx.x & 31;
  int mt  = (w & 3) * 16;
  int ntb = (w >> 2) * 4;
  int col = lane & 15;

  // ---- one-time async stage of Whh slice: local row l (=q*32+within) <- Whh[q*512+j0+within][:]
  {
    unsigned base = (unsigned)(uintptr_t)lds_w;
#pragma unroll 4
    for (int it = 0; it < 32; ++it){
      int cidx = it * 256 + threadIdx.x;          // 8192 chunks of 16B
      int l  = cidx >> 6;                          // local row 0..127
      int ck = cidx & 63;                          // 16B chunk in row
      int q = l >> 5, within = l & 31;
      const bf16* gsrc = Whh + (size_t)(q * H_ + j0 + within) * H_ + ck * 8;
      unsigned ldst = base + (unsigned)(l * (LDSW_ELEM * 2) + ck * 16);
      asm volatile("global_load_async_to_lds_b128 %0, %1, off"
                   :: "v"(ldst), "v"(gsrc) : "memory");
    }
    asm volatile("s_wait_asynccnt 0x0" ::: "memory");
  }

#pragma unroll
  for (int e = 0; e < 8; ++e){
    int p = e * 256 + threadIdx.x;
    lds_c[p >> 5][p & 31] = c0[j0 + (p & 31)];
  }
  __syncthreads();

  const bf16* browL[4];
  int gcol[4], gn[4];
#pragma unroll
  for (int i = 0; i < 4; ++i){
    int nt = ntb + i;
    int q = nt >> 1, off = (nt & 1) * 16;     // q = gate (i,f,g,o), off within 32-slice
    int lrow = q * 32 + off + col;            // local LDS row == gate-column index
    browL[i] = lds_w + (size_t)lrow * LDSW_ELEM;
    gcol[i]  = lrow;
    gn[i]    = q * H_ + j0 + off + col;       // column in xg
  }
  int rb = mt + ((lane & 16) ? 8 : 0);

  for (int t = 0; t < T_; ++t){
    const bf16*  hprev = hs + (size_t)t * BH;
    const float* xgt   = xg + (size_t)t * B_ * G4;
    const bf16*  arow  = hprev + (size_t)(mt + col) * H_;

    v8f acc[4] = {};
#pragma unroll 4
    for (int kb = 0; kb < H_; kb += 32){
      v16bf a = load_a(arow, kb, lane);
#pragma unroll
      for (int i = 0; i < 4; ++i){
        v16bf b = load_b_lds(browL[i], kb, lane);
        acc[i] = wmma_bf16(a, b, acc[i]);
      }
    }
#pragma unroll
    for (int i = 0; i < 4; ++i){
#pragma unroll
      for (int r = 0; r < 8; ++r){
        int bm = rb + r;
        lds_g[bm][gcol[i]] = acc[i][r] + xgt[(size_t)bm * G4 + gn[i]];
      }
    }
    __syncthreads();

    bf16* hcur = hs + (size_t)(t + 1) * BH;
#pragma unroll
    for (int e = 0; e < 8; ++e){
      int p = e * 256 + threadIdx.x;
      int bm = p >> 5, j = p & 31;
      float gi = lds_g[bm][j];
      float gf = lds_g[bm][32 + j];
      float gg = lds_g[bm][64 + j];
      float go = lds_g[bm][96 + j];
      float c = sigm(gf) * lds_c[bm][j] + sigm(gi) * tanh_f(gg);
      lds_c[bm][j] = c;
      hcur[(size_t)bm * H_ + j0 + j] = (bf16)(sigm(go) * tanh_f(c));
    }
    gsync(&bar[0], &bar[1], 32u);
  }
}

// ---------------- phase 3: out = [hs_f ; hs_b] @ Wl^T + bl ----------------
// M = B*T = 16384 (m = b*256 + t), N = 512, K = 1024. grid = 256 m-blocks x 4 n-blocks.
__global__ __launch_bounds__(256) void k_lin(const bf16* __restrict__ hs_f, const bf16* __restrict__ hs_b,
                                             const bf16* __restrict__ Wl, const float* __restrict__ bl,
                                             float* __restrict__ out){
  int m0 = (blockIdx.x >> 2) * 64;
  int n0 = (blockIdx.x & 3) * 128;
  int bq = m0 >> 8;        // batch index (64-row tile never crosses a batch boundary)
  int t0 = m0 & 255;
  int w = threadIdx.x >> 5, lane = threadIdx.x & 31;
  int mt  = (w & 3) * 16;
  int ntb = (w >> 2) * 4;
  int col = lane & 15;

  int tt = t0 + mt + col;  // per-lane A-row time index
  const bf16* af = hs_f + ((size_t)(tt + 1) * B_ + bq) * H_;   // forward slot t+1
  const bf16* ab = hs_b + ((size_t)(T_ - tt) * B_ + bq) * H_;  // backward slot 256-t
  const bf16* brow[4];
#pragma unroll
  for (int i = 0; i < 4; ++i)
    brow[i] = Wl + (size_t)(n0 + (ntb + i) * 16 + col) * (2 * H_);

  v8f acc[4] = {};
#pragma unroll 2
  for (int kb = 0; kb < 2 * H_; kb += 32){
    const bf16* arow = (kb < H_) ? (af + kb) : (ab + (kb - H_));
    v16bf a = load_a(arow, 0, lane);
#pragma unroll
    for (int i = 0; i < 4; ++i){
      v16bf b = load_b(brow[i], kb, lane);
      acc[i] = wmma_bf16(a, b, acc[i]);
    }
  }
  int rb = mt + ((lane & 16) ? 8 : 0);
#pragma unroll
  for (int i = 0; i < 4; ++i){
    int n = n0 + (ntb + i) * 16 + col;
    float bias = bl[n];
#pragma unroll
    for (int r = 0; r < 8; ++r){
      int m = m0 + rb + r;
      out[(size_t)m * O_ + n] = acc[i][r] + bias;
    }
  }
}

// ---------------- host launcher ----------------
extern "C" void kernel_launch(void* const* d_in, const int* in_sizes, int n_in,
                              void* d_out, int out_size, void* d_ws, size_t ws_size,
                              hipStream_t stream){
  (void)in_sizes; (void)n_in; (void)out_size; (void)ws_size;
  const float* X     = (const float*)d_in[0];
  const float* Wih_f = (const float*)d_in[1];
  const float* Whh_f = (const float*)d_in[2];
  const float* bih_f = (const float*)d_in[3];
  const float* bhh_f = (const float*)d_in[4];
  const float* h0_f  = (const float*)d_in[5];
  const float* c0_f  = (const float*)d_in[6];
  const float* Wih_b = (const float*)d_in[7];
  const float* Whh_b = (const float*)d_in[8];
  const float* bih_b = (const float*)d_in[9];
  const float* bhh_b = (const float*)d_in[10];
  const float* h0_b  = (const float*)d_in[11];
  const float* c0_b  = (const float*)d_in[12];
  const float* Wl    = (const float*)d_in[13];
  const float* bl    = (const float*)d_in[14];
  float* out = (float*)d_out;

  char* p = (char*)d_ws;
  auto take = [&](size_t bytes){ char* r = p; p += (bytes + 255) & ~(size_t)255; return r; };
  bf16*  Xbf   = (bf16*)take((size_t)B_ * T_ * I_ * 2);
  bf16*  Wihfb = (bf16*)take((size_t)G4 * I_ * 2);
  bf16*  Whhfb = (bf16*)take((size_t)G4 * H_ * 2);
  bf16*  Wihbb = (bf16*)take((size_t)G4 * I_ * 2);
  bf16*  Whhbb = (bf16*)take((size_t)G4 * H_ * 2);
  bf16*  Wlb   = (bf16*)take((size_t)O_ * 2 * H_ * 2);
  bf16*  hsf   = (bf16*)take((size_t)(T_ + 1) * BH * 2);
  bf16*  hsb   = (bf16*)take((size_t)(T_ + 1) * BH * 2);
  float* xgf   = (float*)take((size_t)T_ * B_ * G4 * 4);
  float* xgb   = (float*)take((size_t)T_ * B_ * G4 * 4);
  unsigned* bar = (unsigned*)take(256);

  int nX = B_ * T_ * I_;
  k_cvt<<<(nX + 255) / 256, 256, 0, stream>>>(X, Xbf, nX);
  int nW = G4 * I_;
  k_cvt<<<(nW + 255) / 256, 256, 0, stream>>>(Wih_f, Wihfb, nW);
  k_cvt<<<(nW + 255) / 256, 256, 0, stream>>>(Whh_f, Whhfb, nW);
  k_cvt<<<(nW + 255) / 256, 256, 0, stream>>>(Wih_b, Wihbb, nW);
  k_cvt<<<(nW + 255) / 256, 256, 0, stream>>>(Whh_b, Whhbb, nW);
  int nWl = O_ * 2 * H_;
  k_cvt<<<(nWl + 255) / 256, 256, 0, stream>>>(Wl, Wlb, nWl);
  k_h0<<<BH / 256, 256, 0, stream>>>(h0_f, h0_b, hsf, hsb);
  k_bar<<<1, 64, 0, stream>>>(bar);

  k_xg<<<T_ * 16, 256, 0, stream>>>(Xbf, Wihfb, bih_f, bhh_f, xgf, 0);
  k_xg<<<T_ * 16, 256, 0, stream>>>(Xbf, Wihbb, bih_b, bhh_b, xgb, 1);

  k_scan<<<32, 256, 0, stream>>>(Whhfb, Whhbb, xgf, xgb, c0_f, c0_b, hsf, hsb, bar);

  k_lin<<<1024, 256, 0, stream>>>(hsf, hsb, Wlb, bl, out);
}
